// MultiHeadAttention_40699110097732
// MI455X (gfx1250) — matrix operand order repaired and checked by reading, not compiled, not run on hardware
//
#include <hip/hip_runtime.h>

// ---------------------------------------------------------------------------
// Multi-head attention for MI455X (gfx1250), wave32 + WMMA bf16 + TDM path.
// B=4, S=2048, D=1024, H=16, HD=64.
// ---------------------------------------------------------------------------

typedef __attribute__((ext_vector_type(16))) __bf16 v16bf;
typedef __attribute__((ext_vector_type(8)))  float  v8f;
typedef unsigned int u32x4 __attribute__((ext_vector_type(4)));
typedef int          i32x8 __attribute__((ext_vector_type(8)));
typedef int          i32x4 __attribute__((ext_vector_type(4)));

#define B_  4
#define S_  2048
#define D_  1024
#define H_  16
#define HD_ 64
#define MS_ (B_ * S_)   // 8192 rows

#ifndef __has_builtin
#define __has_builtin(x) 0
#endif
#if __has_builtin(__builtin_amdgcn_tensor_load_to_lds) && \
    __has_builtin(__builtin_amdgcn_s_wait_tensorcnt)
#define USE_TDM 1
#else
#define USE_TDM 0
#endif

// A-matrix (16x32, 16-bit) element -> K index for this lane half.
// lanes 0-15: VGPR0-3 K=0..7, VGPR4-7 K=16..23 ; lanes 16-31: K=8..15 / 24..31
__device__ __forceinline__ int a_kidx(int half, int e) {
    return (e < 8) ? (half * 8 + e) : (16 + half * 8 + (e - 8));
}
// B-matrix (32x16, 16-bit) element -> K index (lane = column).
__device__ __forceinline__ int b_kidx(int half, int e) {
    return (e < 8) ? (half * 8 + e) : (16 + half * 8 + (e - 8));
}

#if USE_TDM
// LDS byte offset of a generic pointer into shared memory.
__device__ __forceinline__ unsigned lds_off_of(const void* p) {
    return (unsigned)(unsigned long long)
        (__attribute__((address_space(3))) const void*)p;
}

// TDM: DMA a 64x64 bf16 tile (row stride D_ elements in memory) into LDS,
// injecting 4 DWORDs of padding every 32 DWORDs -> LDS row stride 144B
// (72 bf16, 16B aligned, bank-conflict relief).  D# per ISA 08 §8.3/8.4.
// This toolchain's builtin is the 6-arg form:
//   (u32x4 g0, i32x8 g1, i32x4 g2, i32x4 g3, i32x8 extra, i32 cpol)
__device__ __forceinline__ void tdm_load_tile_64x64(const __bf16* gptr,
                                                    unsigned lds_off) {
    unsigned long long ga = (unsigned long long)(const void*)gptr;
    u32x4 g0;
    g0[0] = 1u;                                   // count=1, user descriptor
    g0[1] = lds_off;                              // lds_addr (bytes)
    g0[2] = (unsigned)(ga & 0xFFFFFFFFu);         // global_addr[31:0]
    g0[3] = (unsigned)((ga >> 32) & 0x01FFFFFFu)  // global_addr[56:32]
            | (2u << 30);                         // type=2 ("image")
    i32x8 g1;
    g1[0] = (1 << 16)      // data_size = 2 bytes
          | (1 << 20)      // pad_enable
          | (4 << 22)      // pad_interval: 32 DWORDs (=128B = one 64-elem row)
          | (3 << 25);     // pad_amount: 4 DWORDs (=16B) -> 144B row stride
    g1[1] = (64 << 16);    // tensor_dim0 = 64 (lo16)
    g1[2] = (64 << 16);    // tensor_dim0 hi = 0 | tensor_dim1 lo = 64
    g1[3] = (64 << 16);    // tensor_dim1 hi = 0 | tile_dim0 = 64
    g1[4] = 64;            // tile_dim1 = 64 | tile_dim2 = 0
    g1[5] = D_;            // tensor_dim0_stride = 1024 elements
    g1[6] = 0;             // stride0 hi | stride1 lo (unused, 2D)
    g1[7] = 0;
    i32x4 z4 = {};
    i32x8 z8 = {};
    __builtin_amdgcn_tensor_load_to_lds(g0, g1, z4, z4, z8, 0);
}
#endif

// ---------------------------------------------------------------------------
// Tiled GEMM: C[M x N] = A[M x Kd] * W[Kd x N]  (A f32 or bf16, W f32)
// 256 threads / 8 waves compute a 64x64 tile via v_wmma_f32_16x16x32_bf16.
// ---------------------------------------------------------------------------
template <typename InT, typename OutT>
__global__ __launch_bounds__(256)
void gemm_bf16_wmma(const InT* __restrict__ A, const float* __restrict__ W,
                    OutT* __restrict__ C, int M, int N, int Kd) {
    constexpr int BM = 64, BN = 64, BK = 32;
    constexpr int LDA = BK + 8;
    constexpr int LDB = BN + 8;
    __shared__ __bf16 As[BM * LDA];
    __shared__ __bf16 Bs[BK * LDB];

    const int tid  = threadIdx.x;
    const int lane = tid & 31;
    const int wave = tid >> 5;
    const int half = lane >> 4;
    const int lo   = lane & 15;
    const int m0   = blockIdx.y * BM;
    const int n0   = blockIdx.x * BN;
    const int msub = (wave & 3) * 16;
    const int nsub = (wave >> 2) * 32;
    (void)M;

    v8f acc0 = {};
    v8f acc1 = {};

    for (int kb = 0; kb < Kd; kb += BK) {
#pragma unroll
        for (int i = 0; i < 8; i++) {
            int e = tid * 8 + i;
            int r = e >> 5, c = e & 31;
            As[r * LDA + c] = (__bf16)A[(size_t)(m0 + r) * Kd + kb + c];
        }
#pragma unroll
        for (int i = 0; i < 8; i++) {
            int e = tid * 8 + i;
            int r = e >> 6, c = e & 63;
            Bs[r * LDB + c] = (__bf16)W[(size_t)(kb + r) * N + n0 + c];
        }
        __syncthreads();

        v16bf a, b0, b1;
#pragma unroll
        for (int e = 0; e < 16; e++) a[e]  = As[(msub + lo) * LDA + a_kidx(half, e)];
#pragma unroll
        for (int e = 0; e < 16; e++) b0[e] = Bs[b_kidx(half, e) * LDB + nsub + lo];
#pragma unroll
        for (int e = 0; e < 16; e++) b1[e] = Bs[b_kidx(half, e) * LDB + nsub + 16 + lo];

        acc0 = __builtin_amdgcn_wmma_f32_16x16x32_bf16(false, a, false, b0,
                                                       (short)0, acc0, false, false);
        acc1 = __builtin_amdgcn_wmma_f32_16x16x32_bf16(false, a, false, b1,
                                                       (short)0, acc1, false, false);
        __syncthreads();
    }

#pragma unroll
    for (int r = 0; r < 8; r++) {
        int m = m0 + msub + half * 8 + r;
        C[(size_t)m * N + n0 + nsub + lo]      = (OutT)acc0[r];
        C[(size_t)m * N + n0 + nsub + 16 + lo] = (OutT)acc1[r];
    }
}

// ---------------------------------------------------------------------------
// Flash attention: 128-thread workgroup = 4 waves, each wave owns a 16-query
// tile of one (b, h); the 4 waves share TDM-staged 64x64 K and V tiles in LDS
// per key block.  Online softmax in f32; matmuls via bf16 WMMA.
// ---------------------------------------------------------------------------
__global__ __launch_bounds__(128)
void flash_attn_wmma(const __bf16* __restrict__ Q, const __bf16* __restrict__ K,
                     const __bf16* __restrict__ V,
                     const long long* __restrict__ valid_lens,
                     __bf16* __restrict__ Oa) {
    constexpr int LDV = 72;                 // bf16 row stride (144B, 16B aligned)
    __shared__ __bf16 Klds[64 * LDV];       // 64 keys x 64 dims
    __shared__ __bf16 Vlds[64 * LDV];       // 64 keys x 64 dims
    __shared__ __bf16 Plds[4 * 16 * LDV];   // per-wave 16x64 probability tiles

    const int tid  = threadIdx.x;
    const int lane = tid & 31;
    const int wave = tid >> 5;
    const int half = lane >> 4;
    const int lo   = lane & 15;

    int wg = blockIdx.x;
    const int qb = wg % (S_ / 64); wg /= (S_ / 64);
    const int h  = wg % H_;
    const int b  = wg / H_;
    const int q0 = qb * 64 + wave * 16;
    const int vlen = (int)valid_lens[b];

    const __bf16* Qb = Q + (size_t)b * S_ * D_ + h * HD_;
    const __bf16* Kb = K + (size_t)b * S_ * D_ + h * HD_;
    const __bf16* Vb = V + (size_t)b * S_ * D_ + h * HD_;
    __bf16* Pw = &Plds[wave * 16 * LDV];

    // Q fragments: 16x64 = two K-steps of 32
    v16bf aq0, aq1;
#pragma unroll
    for (int e = 0; e < 16; e++) {
        aq0[e] = Qb[(size_t)(q0 + lo) * D_ + a_kidx(half, e)];
        aq1[e] = Qb[(size_t)(q0 + lo) * D_ + 32 + a_kidx(half, e)];
    }

    v8f o[4] = {};
    float mi[8], li[8];
#pragma unroll
    for (int r = 0; r < 8; r++) { mi[r] = -3.0e38f; li[r] = 0.0f; }

    const float scale = 0.125f;        // 1/sqrt(HD)
    const float NEGF  = -1000000.0f;   // reference mask fill

    for (int key0 = 0; key0 < S_; key0 += 64) {
        // L2 hint for the following block while we work on this one
        int nk = (key0 + 64 < S_) ? key0 + 64 : key0;
        __builtin_prefetch(&Kb[(size_t)nk * D_], 0, 1);
        __builtin_prefetch(&Vb[(size_t)nk * D_], 0, 1);

#if USE_TDM
        if (wave == 0) {
            tdm_load_tile_64x64(&Kb[(size_t)key0 * D_], lds_off_of(Klds));
            tdm_load_tile_64x64(&Vb[(size_t)key0 * D_], lds_off_of(Vlds));
            __builtin_amdgcn_s_wait_tensorcnt(0);
        }
#else
        // cooperative staging fallback: 4096 bf16 per tile, 32 per thread
#pragma unroll
        for (int i = 0; i < 32; i++) {
            int e = tid * 32 + i;
            int r = e >> 6, c = e & 63;
            Klds[r * LDV + c] = Kb[(size_t)(key0 + r) * D_ + c];
            Vlds[r * LDV + c] = Vb[(size_t)(key0 + r) * D_ + c];
        }
#endif
        __syncthreads();

        // ---- scores: S_tile[16 x 64] = Q(16x64) . K^T(64x64) * scale ----
        v8f sc[4];
#pragma unroll
        for (int t = 0; t < 4; t++) {
            const int key = key0 + t * 16 + lo;   // this lane's key column
            v16bf bk0, bk1;
#pragma unroll
            for (int e = 0; e < 16; e++) {
                bk0[e] = Klds[(t * 16 + lo) * LDV + b_kidx(half, e)];
                bk1[e] = Klds[(t * 16 + lo) * LDV + 32 + b_kidx(half, e)];
            }
            v8f s = {};
            s = __builtin_amdgcn_wmma_f32_16x16x32_bf16(false, aq0, false, bk0,
                                                        (short)0, s, false, false);
            s = __builtin_amdgcn_wmma_f32_16x16x32_bf16(false, aq1, false, bk1,
                                                        (short)0, s, false, false);
            const bool masked = (key >= vlen);
#pragma unroll
            for (int r = 0; r < 8; r++) sc[t][r] = masked ? NEGF : s[r] * scale;
        }

        // ---- online softmax row statistics (rows live in half-wave groups) --
        float mnew[8], corr[8];
#pragma unroll
        for (int r = 0; r < 8; r++) {
            float v = fmaxf(fmaxf(sc[0][r], sc[1][r]), fmaxf(sc[2][r], sc[3][r]));
#pragma unroll
            for (int mk = 1; mk < 16; mk <<= 1) v = fmaxf(v, __shfl_xor(v, mk, 32));
            mnew[r] = fmaxf(mi[r], v);
            corr[r] = __expf(mi[r] - mnew[r]);
            mi[r]   = mnew[r];
        }

        // ---- probabilities -> LDS (C-layout -> row-major bf16), row sums ---
#pragma unroll
        for (int r = 0; r < 8; r++) {
            float ps = 0.0f;
#pragma unroll
            for (int t = 0; t < 4; t++) {
                float p = __expf(sc[t][r] - mnew[r]);
                ps += p;
                Pw[(half * 8 + r) * LDV + t * 16 + lo] = (__bf16)p;
            }
#pragma unroll
            for (int mk = 1; mk < 16; mk <<= 1) ps += __shfl_xor(ps, mk, 32);
            li[r] = li[r] * corr[r] + ps;
        }
#pragma unroll
        for (int j = 0; j < 4; j++)
#pragma unroll
            for (int r = 0; r < 8; r++) o[j][r] *= corr[r];

        // per-wave private P region: drain DS stores before A-layout re-read
        asm volatile("s_wait_dscnt 0" ::: "memory");

        v16bf ap0, ap1;
#pragma unroll
        for (int e = 0; e < 16; e++) {
            ap0[e] = Pw[lo * LDV + a_kidx(half, e)];
            ap1[e] = Pw[lo * LDV + 32 + a_kidx(half, e)];
        }

        // ---- O += P(16x64) . V(64x64)  (V served from LDS) ----
#pragma unroll
        for (int j = 0; j < 4; j++) {
            v16bf bv0, bv1;
#pragma unroll
            for (int e = 0; e < 16; e++) {
                bv0[e] = Vlds[b_kidx(half, e) * LDV + j * 16 + lo];
                bv1[e] = Vlds[(32 + b_kidx(half, e)) * LDV + j * 16 + lo];
            }
            o[j] = __builtin_amdgcn_wmma_f32_16x16x32_bf16(false, ap0, false, bv0,
                                                           (short)0, o[j], false, false);
            o[j] = __builtin_amdgcn_wmma_f32_16x16x32_bf16(false, ap1, false, bv1,
                                                           (short)0, o[j], false, false);
        }
        __syncthreads();   // protect K/V tiles before next block's staging
    }

    // ---- normalize and store merged-head output [B*S, D] as bf16 ----
#pragma unroll
    for (int j = 0; j < 4; j++)
#pragma unroll
        for (int r = 0; r < 8; r++) {
            float ov = o[j][r] / li[r];
            int m = q0 + half * 8 + r;
            Oa[(size_t)(b * S_ + m) * D_ + h * HD_ + j * 16 + lo] = (__bf16)ov;
        }
}

// ---------------------------------------------------------------------------
// Launch: 3 projection GEMMs (f32 -> bf16), flash attention, output GEMM.
// Workspace: Qb | Kb | Vb | AttnOut, each 8192x1024 bf16 (16 MB) = 64 MB.
// ---------------------------------------------------------------------------
extern "C" void kernel_launch(void* const* d_in, const int* in_sizes, int n_in,
                              void* d_out, int out_size, void* d_ws, size_t ws_size,
                              hipStream_t stream) {
    (void)in_sizes; (void)n_in; (void)out_size; (void)ws_size;

    const float* queries = (const float*)d_in[0];
    const float* keys    = (const float*)d_in[1];
    const float* values  = (const float*)d_in[2];
    const long long* vl  = (const long long*)d_in[3];   // int64 in reference
    const float* Wq      = (const float*)d_in[4];
    const float* Wk      = (const float*)d_in[5];
    const float* Wv      = (const float*)d_in[6];
    const float* Wo      = (const float*)d_in[7];
    float* out           = (float*)d_out;

    const size_t chunk = (size_t)MS_ * D_;   // 8192*1024 elements
    __bf16* Qb = (__bf16*)d_ws;
    __bf16* Kb = Qb + chunk;
    __bf16* Vb = Kb + chunk;
    __bf16* Ab = Vb + chunk;

    dim3 ggrid(D_ / 64, MS_ / 64);   // (16, 128)
    dim3 gblk(256);

    gemm_bf16_wmma<float, __bf16><<<ggrid, gblk, 0, stream>>>(queries, Wq, Qb, MS_, D_, D_);
    gemm_bf16_wmma<float, __bf16><<<ggrid, gblk, 0, stream>>>(keys,    Wk, Kb, MS_, D_, D_);
    gemm_bf16_wmma<float, __bf16><<<ggrid, gblk, 0, stream>>>(values,  Wv, Vb, MS_, D_, D_);

    dim3 fgrid(B_ * H_ * (S_ / 64));  // 2048 workgroups of 4 waves
    dim3 fblk(128);
    flash_attn_wmma<<<fgrid, fblk, 0, stream>>>(Qb, Kb, Vb, vl, Ab);

    gemm_bf16_wmma<__bf16, float><<<ggrid, gblk, 0, stream>>>(Ab, Wo, out, MS_, D_, D_);
}
